// MargGAN_44246753083785
// MI455X (gfx1250) — matrix operand order repaired and checked by reading, not compiled, not run on hardware
//
#include <hip/hip_runtime.h>
#include <math.h>

#define DP    1024      // padded one-hot dim (real 1000)
#define DREAL 1000
#define BSZ   1024
#define HS    1024
#define KQ    100000
#define CCOLS 50
#define NC    20
#define QBLOCKS (KQ / 8) // 12500 blocks of 8 waves

typedef __attribute__((ext_vector_type(16))) _Float16 v16h;
typedef __attribute__((ext_vector_type(8)))  _Float16 v8h;
typedef __attribute__((ext_vector_type(8)))  float    v8f;

// ---------------- conversion kernels ----------------

// zh[b*1024 + d] = f16(z[b*1000 + d]) (zero-padded K)
__global__ void k_pack_z(const float* __restrict__ z, _Float16* __restrict__ zh) {
    int id = blockIdx.x * 256 + threadIdx.x;     // 1M threads
    int b = id >> 10, d = id & 1023;
    float v = (d < DREAL) ? z[b * DREAL + d] : 0.f;
    zh[id] = (_Float16)v;
}

// LDS-tiled transpose+convert: Bt[n*DP + k] = f16(W[k*Ncols + n]), zero-padded.
// Coalesced on both the f32 read (n contiguous) and the f16 write (k contiguous).
__global__ void k_trans_w(const float* __restrict__ W, _Float16* __restrict__ Bt,
                          int Krows, int Ncols) {
    __shared__ float tile[32][33];               // +1 pad: no LDS bank conflicts
    int kt = blockIdx.x * 32;
    int nt = blockIdx.y * 32;
#pragma unroll
    for (int i = 0; i < 4; ++i) {
        int k = kt + threadIdx.y + i * 8;
        int n = nt + threadIdx.x;
        tile[threadIdx.y + i * 8][threadIdx.x] =
            (k < Krows && n < Ncols) ? W[k * Ncols + n] : 0.f;
    }
    __syncthreads();
#pragma unroll
    for (int i = 0; i < 4; ++i) {
        int n = nt + threadIdx.y + i * 8;
        int k = kt + threadIdx.x;
        Bt[n * DP + k] = (_Float16)tile[threadIdx.x][threadIdx.y + i * 8];
    }
}

// ---------------- WMMA GEMM: C = act(A[1024xDP] * Bt^T + bias) ----------------
// A row-major f16 (lda=DP); Bt holds B^T row-major: Bt[n*DP + k].
// mode 0: ReLU -> f16 out.  mode 1: plain -> f32 out.  Both stride DP.

__device__ __forceinline__ v16h load_afrag(const _Float16* __restrict__ arow,
                                           int kk, int half) {
    // A 16x32 f16 fragment in contiguous k order (same as B):
    //   halfs 0..7  -> K = half*16 + h        (contiguous 8)
    //   halfs 8..15 -> K = half*16 + 8 + h-8  (contiguous 8)
    v8h lo = *(const v8h*)(arow + kk + half * 16);
    v8h hi = *(const v8h*)(arow + kk + half * 16 + 8);
    return __builtin_shufflevector(lo, hi, 0, 1, 2, 3, 4, 5, 6, 7,
                                           8, 9, 10, 11, 12, 13, 14, 15);
}

__global__ void k_gemm_wmma(const _Float16* __restrict__ A,
                            const _Float16* __restrict__ Bt,
                            const float* __restrict__ bias, int Nreal, int mode,
                            _Float16* __restrict__ Cf16, float* __restrict__ Cf32) {
    int lane = threadIdx.x & 31;
    int wave = threadIdx.x >> 5;
    int gw   = blockIdx.x * 8 + wave;     // 1024 waves total (128 blocks * 8)
    int mTile  = gw >> 4;                 // 0..63
    int nGroup = gw & 15;                 // 0..15, each covers 64 columns
    int m0   = mTile * 16;
    int half = lane >> 4;                 // 0: lanes 0-15, 1: lanes 16-31
    int l15  = lane & 15;

    v8f acc[4] = {};
    const _Float16* arow = A + (m0 + l15) * DP;
    // B 32x16 fragment: lane = column n; lanes 0-15 hold K=kk..kk+15,
    // lanes 16-31 hold K=kk+16..kk+31 -> 16 contiguous halfs of Bt row n.
    const _Float16* brow = Bt + (nGroup * 64 + l15) * DP + half * 16;

    // software pipeline: keep next k-step's fragments in flight during WMMAs
    v16h aC = load_afrag(arow, 0, half);
    v16h bC[4];
#pragma unroll
    for (int j = 0; j < 4; ++j) bC[j] = *(const v16h*)(brow + j * 16 * DP);

#pragma unroll 4
    for (int kk = 0; kk < DP; kk += 32) {
        v16h aN = aC;
        v16h bN[4] = {bC[0], bC[1], bC[2], bC[3]};
        int kn = kk + 32;
        if (kn < DP) {
            aN = load_afrag(arow, kn, half);
#pragma unroll
            for (int j = 0; j < 4; ++j)
                bN[j] = *(const v16h*)(brow + j * 16 * DP + kn);
        }
#pragma unroll
        for (int j = 0; j < 4; ++j)
            acc[j] = __builtin_amdgcn_wmma_f32_16x16x32_f16(
                false, aC, false, bC[j], (short)0, acc[j], false, false);
        aC = aN;
#pragma unroll
        for (int j = 0; j < 4; ++j) bC[j] = bN[j];
    }

#pragma unroll
    for (int j = 0; j < 4; ++j) {
        int n = nGroup * 64 + j * 16 + l15;
        float bv = (n < Nreal) ? bias[n] : 0.f;
#pragma unroll
        for (int r = 0; r < 8; ++r) {
            int m = m0 + r + (half << 3);   // C/D layout: VGPR r, M = r (+8 hi lanes)
            float v = acc[j][r] + bv;
            if (mode == 0) {
                v = fmaxf(v, 0.f);
                Cf16[m * DP + n] = (_Float16)v;
            } else {
                Cf32[m * DP + n] = v;
            }
        }
    }
}

// ---------------- per-column log-softmax + clamp, transposed store ----------------
__global__ void k_logsoftmax(const float* __restrict__ out3, float* __restrict__ xT) {
    int id = blockIdx.x * 256 + threadIdx.x;
    if (id >= BSZ * CCOLS) return;
    int b = id / CCOLS, c = id % CCOLS;
    const float* p = out3 + b * DP + c * NC;
    float v[NC];
#pragma unroll
    for (int j = 0; j < NC; ++j) v[j] = p[j];
    float mx = -1e30f;
#pragma unroll
    for (int j = 0; j < NC; ++j) mx = fmaxf(mx, v[j]);
    float s = 0.f;
#pragma unroll
    for (int j = 0; j < NC; ++j) s += expf(v[j] - mx);
    float lse = mx + logf(s);
#pragma unroll
    for (int j = 0; j < NC; ++j) {
        float lp = v[j] - lse;
        lp = fminf(fmaxf(lp, -30.f), 0.f);
        xT[(c * NC + j) * BSZ + b] = lp;   // transposed: batch-contiguous per d
    }
}

// ---------------- extract up to 2 one-hot indices per query row ----------------
__global__ void k_extract(const float* __restrict__ Q, int* __restrict__ idx) {
    int lane = threadIdx.x & 31;
    int k = blockIdx.x * 8 + (threadIdx.x >> 5);
    if (k >= KQ) return;
    const float* row = Q + (size_t)k * DREAL;
    int mn = DREAL, mx = -1;
    for (int d = lane; d < DREAL; d += 32) {
        __builtin_prefetch(row + d + 256, 0, 1);   // global_prefetch_b8, speculative
        if (row[d] != 0.f) { mn = min(mn, d); mx = max(mx, d); }
    }
    for (int off = 16; off >= 1; off >>= 1) {
        mn = min(mn, __shfl_xor(mn, off, 32));
        mx = max(mx, __shfl_xor(mx, off, 32));
    }
    if (lane == 0) {
        if (mx < 0) { mn = 0; mx = -1; }          // cannot happen per setup
        idx[2 * k]     = mn;
        idx[2 * k + 1] = (mx == mn) ? -1 : mx;    // -1: single-hot row (dup index)
    }
}

// ---------------- per-query mean(exp) + weighted squared error ----------------
__global__ void k_query(const float* __restrict__ xT, const int* __restrict__ idx,
                        const float* __restrict__ real_ans,
                        const float* __restrict__ weight,
                        float* __restrict__ partials) {
    __shared__ float wsum[8];
    int lane = threadIdx.x & 31;
    int wv   = threadIdx.x >> 5;
    int k    = blockIdx.x * 8 + wv;
    int i0 = idx[2 * k], i1 = idx[2 * k + 1];
    const float* r0 = xT + i0 * BSZ;
    const float* r1 = (i1 >= 0) ? (xT + i1 * BSZ) : nullptr;
    float s = 0.f;
    if (r1) {
        for (int b = lane; b < BSZ; b += 32) s += expf(r0[b] + r1[b]);
    } else {
        for (int b = lane; b < BSZ; b += 32) s += expf(r0[b]);
    }
    for (int off = 16; off >= 1; off >>= 1) s += __shfl_xor(s, off, 32);
    if (lane == 0) {
        float syn = s * (1.f / (float)BSZ);
        float d = syn - real_ans[k];
        wsum[wv] = weight[k] * d * d;
    }
    __syncthreads();
    if (threadIdx.x == 0) {
        float t = 0.f;
        for (int i = 0; i < 8; ++i) t += wsum[i];   // fixed order: deterministic
        partials[blockIdx.x] = t;
    }
}

// ---------------- deterministic final reduction ----------------
__global__ void k_final(const float* __restrict__ partials, int n, float* __restrict__ out) {
    __shared__ float red[256];
    float s = 0.f;
    for (int i = threadIdx.x; i < n; i += 256) s += partials[i];
    red[threadIdx.x] = s;
    __syncthreads();
    for (int off = 128; off >= 1; off >>= 1) {
        if (threadIdx.x < off) red[threadIdx.x] += red[threadIdx.x + off];
        __syncthreads();
    }
    if (threadIdx.x == 0) out[0] = red[0];
}

// ---------------- launch ----------------
extern "C" void kernel_launch(void* const* d_in, const int* in_sizes, int n_in,
                              void* d_out, int out_size, void* d_ws, size_t ws_size,
                              hipStream_t stream) {
    const float* z    = (const float*)d_in[0];
    const float* W1   = (const float*)d_in[1];
    const float* b1   = (const float*)d_in[2];
    const float* W2   = (const float*)d_in[3];
    const float* b2   = (const float*)d_in[4];
    const float* W3   = (const float*)d_in[5];
    const float* b3   = (const float*)d_in[6];
    const float* Q    = (const float*)d_in[7];
    const float* real = (const float*)d_in[8];
    const float* wt   = (const float*)d_in[9];

    char* ws = (char*)d_ws;
    const size_t MB = 1u << 20;
    _Float16* zh   = (_Float16*)(ws + 0 * MB);    // 2 MB
    _Float16* B1t  = (_Float16*)(ws + 2 * MB);    // 2 MB
    _Float16* B2t  = (_Float16*)(ws + 4 * MB);    // 2 MB
    _Float16* B3t  = (_Float16*)(ws + 6 * MB);    // 2 MB
    _Float16* h1   = (_Float16*)(ws + 8 * MB);    // 2 MB
    _Float16* h2   = (_Float16*)(ws + 10 * MB);   // 2 MB
    float*    out3 = (float*)   (ws + 12 * MB);   // 4 MB (stride DP)
    float*    xT   = (float*)   (ws + 16 * MB);   // 1000*1024*4 B
    int*      idx  = (int*)     (ws + 20 * MB);   // 800 KB
    float*    part = (float*)   (ws + 21 * MB);   // 50 KB

    // 1) convert / pad / transpose to f16 (tiled, coalesced)
    k_pack_z<<<4096, 256, 0, stream>>>(z, zh);
    dim3 tb(32, 8), tg(32, 32);
    k_trans_w<<<tg, tb, 0, stream>>>(W1, B1t, DREAL, HS);    // (1000,1024)
    k_trans_w<<<tg, tb, 0, stream>>>(W2, B2t, HS, HS);       // (1024,1024)
    k_trans_w<<<tg, tb, 0, stream>>>(W3, B3t, HS, DREAL);    // (1024,1000) pad N

    // 2) MLP via WMMA (register double-buffered pipeline)
    k_gemm_wmma<<<128, 256, 0, stream>>>(zh, B1t, b1, HS,    0, h1, nullptr);
    k_gemm_wmma<<<128, 256, 0, stream>>>(h1, B2t, b2, HS,    0, h2, nullptr);
    k_gemm_wmma<<<128, 256, 0, stream>>>(h2, B3t, b3, DREAL, 1, nullptr, out3);

    // 3) per-column log-softmax + clamp, transposed
    k_logsoftmax<<<(BSZ * CCOLS + 255) / 256, 256, 0, stream>>>(out3, xT);

    // 4) sparse index extraction from Q_mask (the 400 MB streaming read)
    k_extract<<<QBLOCKS, 256, 0, stream>>>(Q, idx);

    // 5) per-query gather + exp-mean + weighted SSE partials
    k_query<<<QBLOCKS, 256, 0, stream>>>(xT, idx, real, wt, part);

    // 6) deterministic final reduction
    k_final<<<1, 256, 0, stream>>>(part, QBLOCKS, (float*)d_out);
}
